// CZT_72533407694901
// MI455X (gfx1250) — matrix operand order repaired and checked by reading, not compiled, not run on hardware
//
#include <hip/hip_runtime.h>

// Problem constants
#define MHALF 1840
#define MTOT  3680
#define KTOT  512
#define NTOT  8192
#define WCOLS 1024

typedef float v2f __attribute__((ext_vector_type(2)));
typedef float v8f __attribute__((ext_vector_type(8)));
typedef __attribute__((ext_vector_type(16))) _Float16 v16h;
typedef _Float16 f16;

__device__ __forceinline__ float hclamp(float v) {
    return fminf(1.0f, fmaxf(-1.0f, v));
}

// ---------------------------------------------------------------------------
// Output 1: W = clip([[Wr, -Wi],[Wi, Wr]], -1, 1)   [3680 x 1024]
// ---------------------------------------------------------------------------
__global__ __launch_bounds__(256) void czt_wout(const float* __restrict__ Wr,
                                                const float* __restrict__ Wi,
                                                float* __restrict__ out) {
    unsigned t = blockIdx.x * 256u + threadIdx.x;
    int j = t >> 8;
    int i = (t & 255) * 4;
    const float4* src;
    float sgn = 1.0f;
    if (j < MHALF) {
        if (i < KTOT) { src = (const float4*)(Wr + (size_t)j * KTOT + i); }
        else          { src = (const float4*)(Wi + (size_t)j * KTOT + (i - KTOT)); sgn = -1.0f; }
    } else {
        int jj = j - MHALF;
        if (i < KTOT) { src = (const float4*)(Wi + (size_t)jj * KTOT + i); }
        else          { src = (const float4*)(Wr + (size_t)jj * KTOT + (i - KTOT)); }
    }
    float4 v = *src;
    v.x = hclamp(sgn * v.x); v.y = hclamp(sgn * v.y);
    v.z = hclamp(sgn * v.z); v.w = hclamp(sgn * v.w);
    *(float4*)(out + (size_t)t * 4) = v;
}

// ---------------------------------------------------------------------------
// convA: fused U/V construction + f16 hi/lo split, K-blocked layout
//   Ah/Al[(k>>5)][m][k&31]   grid(16, 115)
// ---------------------------------------------------------------------------
__global__ __launch_bounds__(256) void czt_convA(const float* __restrict__ Wr,
                                                 const float* __restrict__ Wi,
                                                 const float* __restrict__ ac,
                                                 const float* __restrict__ as_,
                                                 f16* __restrict__ Ah,
                                                 f16* __restrict__ Al) {
    int t = threadIdx.x;
    int m0 = blockIdx.y * 32, k0 = blockIdx.x * 32;
    int mr = m0 + (t >> 3);
    int kk = (t & 7) * 4;
    bool real = mr < MHALF;
    int m = real ? mr : mr - MHALF;
    const float* pr = Wr + (size_t)m * KTOT;
    const float* pi = Wi + (size_t)m * KTOT;
    union { f16 h[4]; uint2 u; } H, L;
    #pragma unroll
    for (int j = 0; j < 4; j++) {
        int k = k0 + kk + j;
        float u = real ? (pr[k] * ac[k] - pi[k] * as_[k])
                       : (pi[k] * ac[k] + pr[k] * as_[k]);
        f16 hh = (f16)u;
        H.h[j] = hh;
        L.h[j] = (f16)(u - (float)hh);
    }
    size_t off = ((size_t)blockIdx.x * MTOT + mr) * 32 + kk;
    *(uint2*)(Ah + off) = H.u;
    *(uint2*)(Al + off) = L.u;
}

// ---------------------------------------------------------------------------
// convB: x[k][n] -> Bh/Bl[(k>>5)][n][k&31] (transposed, K-blocked) via LDS
//   grid(128, 16): 64n x 32k tiles
// ---------------------------------------------------------------------------
__global__ __launch_bounds__(256) void czt_convB(const float* __restrict__ x,
                                                 f16* __restrict__ Bh,
                                                 f16* __restrict__ Bl) {
    __shared__ float tile[32][68];
    int t = threadIdx.x;
    int n0 = blockIdx.x * 64, k0 = blockIdx.y * 32;
    {
        int r = t >> 3;
        int c = (t & 7) * 8;
        const float* src = x + (size_t)(k0 + r) * NTOT + n0 + c;
        *(float4*)&tile[r][c]     = *(const float4*)src;
        *(float4*)&tile[r][c + 4] = *(const float4*)(src + 4);
    }
    __syncthreads();
    int n  = t >> 2;
    int pc = (t & 3) * 8;
    union { f16 h[8]; uint4 u; } H, L;
    #pragma unroll
    for (int j = 0; j < 8; j++) {
        float u = tile[pc + j][n];
        f16 hh = (f16)u;
        H.h[j] = hh;
        L.h[j] = (f16)(u - (float)hh);
    }
    size_t off = ((size_t)blockIdx.y * NTOT + n0 + n) * 32 + pc;
    *(uint4*)(Bh + off) = H.u;
    *(uint4*)(Bl + off) = L.u;
}

// ---------------------------------------------------------------------------
// Split-f16 GEMM: X = (Ah+Al)@(Bh+Bl) dropping Al*Bl, f32 accumulate.
// V_WMMA_F32_16X16X32_F16. 128 threads (4 waves), 2x2 tile/wave,
// LDS double-buffered (51.2 KB), no persistent staging registers.
// ---------------------------------------------------------------------------
#define FBM 32
#define FBN 128
#define FPITCH 40               // f16 pitch: 80B rows, 16B-aligned pieces
#define FNCHUNK (KTOT / 32)     // 16

__device__ __forceinline__ v16h load_frag(const f16* p0, const f16* p1) {
    v16h r;
    ((uint4*)&r)[0] = *(const uint4*)p0;
    ((uint4*)&r)[1] = *(const uint4*)p1;
    return r;
}

__global__ __launch_bounds__(128) void czt_gemm_f16(const f16* __restrict__ Ah,
                                                    const f16* __restrict__ Al,
                                                    const f16* __restrict__ Bh,
                                                    const f16* __restrict__ Bl,
                                                    float* __restrict__ X) {
    __shared__ f16 sA[2][2][FBM * FPITCH];   // [buf][hi/lo]  10,240 B
    __shared__ f16 sB[2][2][FBN * FPITCH];   // [buf][hi/lo]  40,960 B

    const int tid  = threadIdx.x;            // 0..127
    const int wave = tid >> 5;               // 0..3
    const int lane = tid & 31;
    const int m0 = blockIdx.y * FBM;
    const int n0 = blockIdx.x * FBN;

    // staging coords (128 threads)
    const int srow = tid >> 2;               // 0..31  (A row / B n base)
    const int spc  = (tid & 3) * 8;          // 16B piece
    const f16* gAh = Ah + ((size_t)(m0 + srow)) * 32 + spc;
    const f16* gAl = Al + ((size_t)(m0 + srow)) * 32 + spc;
    const f16* gBh = Bh + ((size_t)(n0 + srow)) * 32 + spc;
    const f16* gBl = Bl + ((size_t)(n0 + srow)) * 32 + spc;
    const size_t strideAc = (size_t)MTOT * 32;
    const size_t strideBc = (size_t)NTOT * 32;

    auto stage = [&](int chunk, int buf) {
        *(uint4*)&sA[buf][0][srow * FPITCH + spc] =
            *(const uint4*)(gAh + (size_t)chunk * strideAc);
        *(uint4*)&sA[buf][1][srow * FPITCH + spc] =
            *(const uint4*)(gAl + (size_t)chunk * strideAc);
        #pragma unroll
        for (int sub = 0; sub < 4; sub++) {
            *(uint4*)&sB[buf][0][(srow + sub * 32) * FPITCH + spc] =
                *(const uint4*)(gBh + (size_t)chunk * strideBc + (size_t)sub * 32 * 32);
            *(uint4*)&sB[buf][1][(srow + sub * 32) * FPITCH + spc] =
                *(const uint4*)(gBl + (size_t)chunk * strideBc + (size_t)sub * 32 * 32);
        }
    };

    v8f acc00 = {}, acc01 = {}, acc10 = {}, acc11 = {};
    const int ln   = lane & 15;
    const int half = lane >> 4;
    const int nb   = wave * 32;

    stage(0, 0);
    __syncthreads();

    #define WM(ACC, A, B) ACC = __builtin_amdgcn_wmma_f32_16x16x32_f16( \
        false, A, false, B, (short)0, ACC, false, false)

    for (int i = 0; i < FNCHUNK; i++) {
        if (i + 1 < FNCHUNK) stage(i + 1, (i + 1) & 1);

        const f16* a_h = sA[i & 1][0];
        const f16* a_l = sA[i & 1][1];
        const f16* b_h = sB[i & 1][0];
        const f16* b_l = sB[i & 1][1];

        // A frag: lanes 0-15: K 0-7 & 16-23; lanes 16-31: K 8-15 & 24-31
        const int ra0 = ln * FPITCH + half * 8;
        const int ra1 = (16 + ln) * FPITCH + half * 8;
        // B frag: lanes 0-15: K 0-7 & 16-23; lanes 16-31: K 8-15 & 24-31
        const int rb0 = (nb + ln) * FPITCH + half * 8;
        const int rb1 = (nb + 16 + ln) * FPITCH + half * 8;

        v16h fAh0 = load_frag(a_h + ra0, a_h + ra0 + 16);
        v16h fAh1 = load_frag(a_h + ra1, a_h + ra1 + 16);
        v16h fAl0 = load_frag(a_l + ra0, a_l + ra0 + 16);
        v16h fAl1 = load_frag(a_l + ra1, a_l + ra1 + 16);
        v16h fBh0 = load_frag(b_h + rb0, b_h + rb0 + 16);
        v16h fBh1 = load_frag(b_h + rb1, b_h + rb1 + 16);

        // hi*hi and lo*hi terms (accumulator reuse distance = 4)
        WM(acc00, fAh0, fBh0); WM(acc10, fAh1, fBh0);
        WM(acc01, fAh0, fBh1); WM(acc11, fAh1, fBh1);
        WM(acc00, fAl0, fBh0); WM(acc10, fAl1, fBh0);
        WM(acc01, fAl0, fBh1); WM(acc11, fAl1, fBh1);

        // hi*lo terms (Bl fragments loaded late to limit liveness)
        v16h fBl0 = load_frag(b_l + rb0, b_l + rb0 + 16);
        v16h fBl1 = load_frag(b_l + rb1, b_l + rb1 + 16);
        WM(acc00, fAh0, fBl0); WM(acc10, fAh1, fBl0);
        WM(acc01, fAh0, fBl1); WM(acc11, fAh1, fBl1);

        __syncthreads();
    }
    #undef WM

    const int rbase = half * 8;
    float* op = X + (size_t)m0 * NTOT + n0 + nb + ln;
    #pragma unroll
    for (int r = 0; r < 8; r++) {
        int row = rbase + r;
        op[(size_t)row * NTOT]             = acc00[r];
        op[(size_t)row * NTOT + 16]        = acc01[r];
        op[(size_t)(row + 16) * NTOT]      = acc10[r];
        op[(size_t)(row + 16) * NTOT + 16] = acc11[r];
    }
}

// ---------------------------------------------------------------------------
// Fallback fp32 WMMA GEMM (used only if d_ws is too small)
// ---------------------------------------------------------------------------
#define BM 32
#define BN 256
#define BK 16
#define APAD 17
#define NCHUNK (KTOT / BK)

__global__ __launch_bounds__(256) void czt_gemm_f32(const float* __restrict__ x,
                                                    const float* __restrict__ Wr,
                                                    const float* __restrict__ Wi,
                                                    const float* __restrict__ ac,
                                                    const float* __restrict__ as_,
                                                    float* __restrict__ X) {
    __shared__ float aT[2][BM * APAD];
    __shared__ float bT[2][BK * BN];

    const int tid  = threadIdx.x;
    const int wave = tid >> 5;
    const int lane = tid & 31;
    const int m0 = blockIdx.y * BM;
    const int n0 = blockIdx.x * BN;

    const bool mixed  = (m0 < MHALF) && (m0 + BM > MHALF);
    const bool isReal = (m0 + BM) <= MHALF;

    const int c_st = tid & 15;
    const int r0   = tid >> 4;
    const int r1   = r0 + 16;
    const float *p00, *p01, *p10, *p11;
    float sgn;
    {
        const float* b0 = isReal ? Wr : Wi;
        const float* b1 = isReal ? Wi : Wr;
        sgn = isReal ? -1.0f : 1.0f;
        int mr0 = m0 + r0 - (isReal ? 0 : MHALF);
        p00 = b0 + (size_t)mr0 * KTOT;
        p01 = b1 + (size_t)mr0 * KTOT;
        p10 = b0 + (size_t)(mr0 + 16) * KTOT;
        p11 = b1 + (size_t)(mr0 + 16) * KTOT;
    }
    const int rB = tid >> 4;
    const int cB = (tid & 15) * 4;
    const float* xrow = x + (size_t)rB * NTOT + n0 + cB;

    auto stage = [&](int kc, int buf) {
        if (!mixed) {
            int k = kc + c_st;
            float s0 = ac[k];
            float s1 = sgn * as_[k];
            aT[buf][r0 * APAD + c_st] = p00[k] * s0 + p01[k] * s1;
            aT[buf][r1 * APAD + c_st] = p10[k] * s0 + p11[k] * s1;
        } else {
            #pragma unroll
            for (int e = tid; e < BM * BK; e += 256) {
                int r = e >> 4, c = e & 15;
                int m = m0 + r, k = kc + c;
                float u;
                if (m < MHALF) u = Wr[m * KTOT + k] * ac[k] - Wi[m * KTOT + k] * as_[k];
                else { int mm = m - MHALF; u = Wi[mm * KTOT + k] * ac[k] + Wr[mm * KTOT + k] * as_[k]; }
                aT[buf][r * APAD + c] = u;
            }
        }
        const float* src = xrow + (size_t)kc * NTOT;
        float* dst = &bT[buf][rB * BN + cB];
        #pragma unroll
        for (int q = 0; q < 4; q++)
            *(float4*)(dst + q * 64) = *(const float4*)(src + q * 64);
    };

    v8f acc00 = {}, acc01 = {}, acc10 = {}, acc11 = {};
    const int ln = lane & 15;
    const int kh = (lane >> 4) << 1;
    const int nb = wave * 32;

    stage(0, 0);
    __syncthreads();
    for (int i = 0; i < NCHUNK; i++) {
        if (i + 1 < NCHUNK) stage((i + 1) * BK, (i + 1) & 1);
        const float* a = aT[i & 1];
        const float* b = bT[i & 1];
        #pragma unroll
        for (int kk = 0; kk < BK; kk += 4) {
            v2f a0, a1, b0, b1;
            a0.x = a[ln * APAD + kk + kh];        a0.y = a[ln * APAD + kk + kh + 1];
            a1.x = a[(16 + ln) * APAD + kk + kh]; a1.y = a[(16 + ln) * APAD + kk + kh + 1];
            b0.x = b[(kk + kh) * BN + nb + ln];      b0.y = b[(kk + kh + 1) * BN + nb + ln];
            b1.x = b[(kk + kh) * BN + nb + 16 + ln]; b1.y = b[(kk + kh + 1) * BN + nb + 16 + ln];
            acc00 = __builtin_amdgcn_wmma_f32_16x16x4_f32(false, a0, false, b0, (short)0, acc00, false, false);
            acc01 = __builtin_amdgcn_wmma_f32_16x16x4_f32(false, a0, false, b1, (short)0, acc01, false, false);
            acc10 = __builtin_amdgcn_wmma_f32_16x16x4_f32(false, a1, false, b0, (short)0, acc10, false, false);
            acc11 = __builtin_amdgcn_wmma_f32_16x16x4_f32(false, a1, false, b1, (short)0, acc11, false, false);
        }
        __syncthreads();
    }
    const int rbase = (lane >> 4) * 8;
    float* op = X + (size_t)m0 * NTOT + n0 + nb + ln;
    #pragma unroll
    for (int r = 0; r < 8; r++) {
        int row = rbase + r;
        op[(size_t)row * NTOT]             = acc00[r];
        op[(size_t)row * NTOT + 16]        = acc01[r];
        op[(size_t)(row + 16) * NTOT]      = acc10[r];
        op[(size_t)(row + 16) * NTOT + 16] = acc11[r];
    }
}

// ---------------------------------------------------------------------------
extern "C" void kernel_launch(void* const* d_in, const int* in_sizes, int n_in,
                              void* d_out, int out_size, void* d_ws, size_t ws_size,
                              hipStream_t stream) {
    (void)in_sizes; (void)n_in; (void)out_size;
    const float* x   = (const float*)d_in[0];
    const float* Wr  = (const float*)d_in[1];
    const float* Wi  = (const float*)d_in[2];
    const float* ac  = (const float*)d_in[3];
    const float* as_ = (const float*)d_in[4];

    float* Wout = (float*)d_out;
    float* Xout = Wout + (size_t)MTOT * WCOLS;

    czt_wout<<<dim3((MTOT * WCOLS / 4) / 256), dim3(256), 0, stream>>>(Wr, Wi, Wout);

    const size_t nA = (size_t)MTOT * KTOT;
    const size_t nB = (size_t)KTOT * NTOT;
    const size_t need = (nA * 2 + nB * 2) * sizeof(f16);   // ~24.3 MB

    if (ws_size >= need) {
        f16* Ah = (f16*)d_ws;
        f16* Al = Ah + nA;
        f16* Bh = Al + nA;
        f16* Bl = Bh + nB;
        czt_convA<<<dim3(KTOT / 32, MTOT / 32), dim3(256), 0, stream>>>(Wr, Wi, ac, as_, Ah, Al);
        czt_convB<<<dim3(NTOT / 64, KTOT / 32), dim3(256), 0, stream>>>(x, Bh, Bl);
        czt_gemm_f16<<<dim3(NTOT / FBN, MTOT / FBM), dim3(128), 0, stream>>>(Ah, Al, Bh, Bl, Xout);
    } else {
        czt_gemm_f32<<<dim3(NTOT / BN, MTOT / BM), dim3(256), 0, stream>>>(x, Wr, Wi, ac, as_, Xout);
    }
}
